// GNNGeneric_18047452578601
// MI455X (gfx1250) — matrix operand order repaired and checked by reading, not compiled, not run on hardware
//
#include <hip/hip_runtime.h>

// ---------------------------------------------------------------------------
// CDNA5 (gfx1250) fused GNN pipeline.
// All GEMMs run on v_wmma_f32_16x16x32_bf16 (wave32, fp32 accumulate).
// Weights are pre-swizzled into WMMA-B bf16 fragment layout, staged once per
// workgroup into LDS (320KB/WGP), and read with ds_load_b128 clauses in the
// hot loop. A fragments are built on the fly from gathered node features
// (gathers hit the 192MB L2: node table is only 10MB).
//
// VGPR layouts (CDNA5 ISA §7.12.2, 16-bit A 16x32 / B 32x16 / f32 C 16x16):
//   A: lane L (row = L&15, kbase = (L>>4)*8): halves 0..7 -> k = kbase+h,
//      halves 8..15 -> k = kbase+16+(h-8)
//   B: lane L (col = L&15): same k order as A: halves 0..7 -> k = kbase+h, 8..15 -> kbase+16+(h-8)
//   C/D: lane L holds col n = L&15; VGPR r holds row m = r + 8*(L>>4)
// ---------------------------------------------------------------------------

typedef __attribute__((ext_vector_type(16))) __bf16    v16bf;
typedef __attribute__((ext_vector_type(8)))  float     v8f;
typedef __attribute__((ext_vector_type(4)))  float     v4f;
typedef __attribute__((ext_vector_type(4)))  unsigned  v4u;

#define ZDIM 128

__device__ __forceinline__ int imin(int a, int b) { return a < b ? a : b; }

// Pack two fp32 -> packed bf16 pair: round-half-up then v_perm_b32 grabs the
// two high halves in a single instruction (1 add per float + 1 perm per pair).
__device__ __forceinline__ unsigned pack2_bf16(float lo, float hi) {
  unsigned a = __builtin_bit_cast(unsigned, lo) + 0x8000u;
  unsigned b = __builtin_bit_cast(unsigned, hi) + 0x8000u;
  // sel 0x07060302: bytes {s1.b2, s1.b3, s0.b2, s0.b3} -> [lo.bf16 | hi.bf16]
  return __builtin_amdgcn_perm(b, a, 0x07060302u);
}

__device__ __forceinline__ v16bf pack16(const float* f) {
  union { v16bf v; unsigned u[8]; } r;
#pragma unroll
  for (int i = 0; i < 8; ++i) r.u[i] = pack2_bf16(f[2 * i], f[2 * i + 1]);
  return r.v;
}

// Cooperative copy of the bf16 weight fragment into LDS (16B per thread-iter).
__device__ __forceinline__ void stage_frag(const unsigned short* __restrict__ g,
                                           unsigned short* __restrict__ lds,
                                           int kw) {
  const int total = (kw * ZDIM) / 8;  // 16-byte units
  for (int i = threadIdx.x; i < total; i += 256)
    *(v4u*)(lds + i * 8) = *(const v4u*)(g + i * 8);
  __syncthreads();
}

// B fragment for (K-chunk c, 16-col tile n) from LDS: contiguous 32B per lane.
__device__ __forceinline__ v16bf lds_bfrag(const unsigned short* __restrict__ lds,
                                           int c, int n, int lane) {
  return *(const v16bf*)(lds + ((size_t)((c * 8 + n) * 32 + lane) << 4));
}

// ---------------------------------------------------------------------------
// Prep: swizzle fp32 weights [Kw x 128] into bf16 WMMA-B fragment layout.
// flat index i = ((c*8 + n)*32 + lane)*16 + h  (RNE conversion here, one-off)
// ---------------------------------------------------------------------------
__global__ void fragize_kernel(const float* __restrict__ W,
                               unsigned short* __restrict__ frag, int Kw) {
  int i = blockIdx.x * blockDim.x + threadIdx.x;
  if (i >= Kw * ZDIM) return;
  int h = i & 15, l = (i >> 4) & 31, n = (i >> 9) & 7, c = i >> 12;
  int k    = c * 32 + (l >> 4) * 8 + (h & 7) + (h >> 3) * 16;
  int ncol = n * 16 + (l & 15);
  unsigned u = __builtin_bit_cast(unsigned, W[(size_t)k * ZDIM + ncol]);
  unsigned r = u + 0x7fffu + ((u >> 16) & 1u);
  frag[i] = (unsigned short)(r >> 16);
}

__global__ void zero_kernel(float* __restrict__ p, int n) {
  int i = blockIdx.x * blockDim.x + threadIdx.x;
  if (i < n) p[i] = 0.f;
}

__global__ void count_kernel(const int* __restrict__ dst, float* __restrict__ cnt, int E) {
  int e = blockIdx.x * blockDim.x + threadIdx.x;
  if (e < E) atomicAdd(cnt + dst[e], 1.0f);
}

// ---------------------------------------------------------------------------
// Edge message kernel: one wave = 16 edges x 128 outputs.
// feat = [x_i | x_j - x_i | x_j * x_i | edge_attr];  msg = relu(feat@Wm + bm)
// 3*D WMMA part = three K-block sub-GEMMs; 4-wide edge_attr tail = fp32 dot4.
// Scatter via global_atomic_add_f32 into per-dst sums (10MB -> L2 resident).
// ---------------------------------------------------------------------------
template <int D>
__global__ void __launch_bounds__(256) edge_msg_kernel(
    const float* __restrict__ x,
    const int* __restrict__ src, const int* __restrict__ dst,
    const float* __restrict__ eattr,
    const unsigned short* __restrict__ WmFrag,   // bf16 frag of Wm rows [0, 3D)
    const float* __restrict__ Wm,                // original fp32 (3D+4) x 128
    const float* __restrict__ bm,
    float* __restrict__ sumb, int E) {
  extern __shared__ unsigned short smem[];
  constexpr int KW = 3 * D;
  stage_frag(WmFrag, smem, KW);                  // uniform: before any exit

  const int lane  = threadIdx.x & 31;
  const int tile  = blockIdx.x * (blockDim.x >> 5) + (threadIdx.x >> 5);
  const int eBase = tile * 16;
  if (eBase >= E) return;  // wave-uniform: EXEC stays all-1s for WMMA
  const int half = lane >> 4, lrow = lane & 15;

  const int eA = imin(eBase + lrow, E - 1);
  const float* xi = x + (size_t)dst[eA] * D;  // x_i = x[target]
  const float* xj = x + (size_t)src[eA] * D;  // x_j = x[source]

  v8f acc[8] = {};

  const int CH = KW / 32;
  for (int c = 0; c < CH; ++c) {
    const int mode = c / (D / 32);                     // 0: x_i, 1: diff, 2: prod
    const int kk   = (c % (D / 32)) * 32 + half * 8;
    float fb[16];
#pragma unroll
    for (int s = 0; s < 2; ++s) {
      const int k0 = kk + s * 16;
      v4f i0 = *(const v4f*)(xi + k0);
      v4f i1 = *(const v4f*)(xi + k0 + 4);
      v4f j0 = *(const v4f*)(xj + k0);
      v4f j1 = *(const v4f*)(xj + k0 + 4);
      float iv[8] = {i0.x, i0.y, i0.z, i0.w, i1.x, i1.y, i1.z, i1.w};
      float jv[8] = {j0.x, j0.y, j0.z, j0.w, j1.x, j1.y, j1.z, j1.w};
#pragma unroll
      for (int t = 0; t < 8; ++t)
        fb[s * 8 + t] = (mode == 0) ? iv[t]
                      : (mode == 1) ? (jv[t] - iv[t])
                                    : (jv[t] * iv[t]);
    }
    v16bf a = pack16(fb);
    v16bf b[8];
#pragma unroll
    for (int n = 0; n < 8; ++n) b[n] = lds_bfrag(smem, c, n, lane);  // ds clause
#pragma unroll
    for (int n = 0; n < 8; ++n)
      acc[n] = __builtin_amdgcn_wmma_f32_16x16x32_bf16(
          false, a, false, b[n], (short)0, acc[n], false, false);
  }

  // Epilogue: edge_attr tail (rows 3D..3D+3 of Wm) + bias + relu + scatter.
  float ea[8][4];
  int dr[8], ev[8];
#pragma unroll
  for (int r = 0; r < 8; ++r) {
    int e = eBase + r + 8 * half;   // C-layout row m = r + 8*half
    ev[r] = e;
    int ec = imin(e, E - 1);
    v4f t = *(const v4f*)(eattr + (size_t)ec * 4);
    ea[r][0] = t.x; ea[r][1] = t.y; ea[r][2] = t.z; ea[r][3] = t.w;
    dr[r] = dst[ec];
  }
#pragma unroll
  for (int n = 0; n < 8; ++n) {
    const int   ncol = n * 16 + lrow;
    const float w0 = Wm[(size_t)(3 * D + 0) * ZDIM + ncol];
    const float w1 = Wm[(size_t)(3 * D + 1) * ZDIM + ncol];
    const float w2 = Wm[(size_t)(3 * D + 2) * ZDIM + ncol];
    const float w3 = Wm[(size_t)(3 * D + 3) * ZDIM + ncol];
    const float bias = bm[ncol];
#pragma unroll
    for (int r = 0; r < 8; ++r) {
      float v = acc[n][r] + bias +
                ea[r][0] * w0 + ea[r][1] * w1 + ea[r][2] * w2 + ea[r][3] * w3;
      v = fmaxf(v, 0.f);
      if (ev[r] < E) atomicAdd(sumb + (size_t)dr[r] * ZDIM + ncol, v);
    }
  }
}

// ---------------------------------------------------------------------------
// Node kernel: agg = sum/max(cnt,1);  out = relu(concat(x, agg) @ Wf + bf)
// One wave = 16 nodes x 128 outputs; K = D + 128 (multiple of 32).
// ---------------------------------------------------------------------------
template <int D>
__global__ void __launch_bounds__(256) node_mlp_kernel(
    const float* __restrict__ x,
    const float* __restrict__ sumb,
    const float* __restrict__ cnt,
    const unsigned short* __restrict__ WfFrag,
    const float* __restrict__ bfv,
    float* __restrict__ outp, int N) {
  extern __shared__ unsigned short smem[];
  constexpr int KW = D + ZDIM;
  stage_frag(WfFrag, smem, KW);

  const int lane  = threadIdx.x & 31;
  const int tile  = blockIdx.x * (blockDim.x >> 5) + (threadIdx.x >> 5);
  const int nBase = tile * 16;
  if (nBase >= N) return;
  const int half = lane >> 4, lrow = lane & 15;
  const int node = imin(nBase + lrow, N - 1);
  const float* xr = x + (size_t)node * D;
  const float* sr = sumb + (size_t)node * ZDIM;
  const float invc = 1.f / fmaxf(cnt[node], 1.f);

  v8f acc[8] = {};
  const int CH = KW / 32;
  for (int c = 0; c < CH; ++c) {
    float fb[16];
#pragma unroll
    for (int s = 0; s < 2; ++s) {
      const int k0 = c * 32 + half * 8 + s * 16;   // 8-float segment, never straddles D
      v4f a0, a1; float scale;
      if (k0 < D) { a0 = *(const v4f*)(xr + k0);       a1 = *(const v4f*)(xr + k0 + 4);       scale = 1.f;  }
      else        { a0 = *(const v4f*)(sr + (k0 - D)); a1 = *(const v4f*)(sr + (k0 - D) + 4); scale = invc; }
      fb[s*8+0]=a0.x*scale; fb[s*8+1]=a0.y*scale; fb[s*8+2]=a0.z*scale; fb[s*8+3]=a0.w*scale;
      fb[s*8+4]=a1.x*scale; fb[s*8+5]=a1.y*scale; fb[s*8+6]=a1.z*scale; fb[s*8+7]=a1.w*scale;
    }
    v16bf a = pack16(fb);
    v16bf b[8];
#pragma unroll
    for (int n = 0; n < 8; ++n) b[n] = lds_bfrag(smem, c, n, lane);
#pragma unroll
    for (int n = 0; n < 8; ++n)
      acc[n] = __builtin_amdgcn_wmma_f32_16x16x32_bf16(
          false, a, false, b[n], (short)0, acc[n], false, false);
  }
#pragma unroll
  for (int n = 0; n < 8; ++n) {
    const int   ncol = n * 16 + lrow;
    const float bias = bfv[ncol];
#pragma unroll
    for (int r = 0; r < 8; ++r) {
      const int nr = nBase + r + 8 * half;
      if (nr < N) outp[(size_t)nr * ZDIM + ncol] = fmaxf(acc[n][r] + bias, 0.f);
    }
  }
}

// ---------------------------------------------------------------------------
// Fusion: out = relu(concat(x1,x2,x3) @ Wl + bl), K = 384.
// ---------------------------------------------------------------------------
__global__ void __launch_bounds__(256) fusion_kernel(
    const float* __restrict__ x1, const float* __restrict__ x2,
    const float* __restrict__ x3,
    const unsigned short* __restrict__ WlFrag, const float* __restrict__ blv,
    float* __restrict__ outp, int N) {
  extern __shared__ unsigned short smem[];
  stage_frag(WlFrag, smem, 3 * ZDIM);

  const int lane  = threadIdx.x & 31;
  const int tile  = blockIdx.x * (blockDim.x >> 5) + (threadIdx.x >> 5);
  const int nBase = tile * 16;
  if (nBase >= N) return;
  const int half = lane >> 4, lrow = lane & 15;
  const int node = imin(nBase + lrow, N - 1);
  const float* p1 = x1 + (size_t)node * ZDIM;
  const float* p2 = x2 + (size_t)node * ZDIM;
  const float* p3 = x3 + (size_t)node * ZDIM;

  v8f acc[8] = {};
  const int CH = (3 * ZDIM) / 32;  // 12
  for (int c = 0; c < CH; ++c) {
    float fb[16];
#pragma unroll
    for (int s = 0; s < 2; ++s) {
      const int k0 = c * 32 + half * 8 + s * 16;
      const float* p = (k0 < ZDIM) ? p1 : (k0 < 2 * ZDIM) ? p2 : p3;
      const int ko = k0 & (ZDIM - 1);
      v4f a0 = *(const v4f*)(p + ko);
      v4f a1 = *(const v4f*)(p + ko + 4);
      fb[s*8+0]=a0.x; fb[s*8+1]=a0.y; fb[s*8+2]=a0.z; fb[s*8+3]=a0.w;
      fb[s*8+4]=a1.x; fb[s*8+5]=a1.y; fb[s*8+6]=a1.z; fb[s*8+7]=a1.w;
    }
    v16bf a = pack16(fb);
    v16bf b[8];
#pragma unroll
    for (int n = 0; n < 8; ++n) b[n] = lds_bfrag(smem, c, n, lane);
#pragma unroll
    for (int n = 0; n < 8; ++n)
      acc[n] = __builtin_amdgcn_wmma_f32_16x16x32_bf16(
          false, a, false, b[n], (short)0, acc[n], false, false);
  }
#pragma unroll
  for (int n = 0; n < 8; ++n) {
    const int   ncol = n * 16 + lrow;
    const float bias = blv[ncol];
#pragma unroll
    for (int r = 0; r < 8; ++r) {
      const int nr = nBase + r + 8 * half;
      if (nr < N) outp[(size_t)nr * ZDIM + ncol] = fmaxf(acc[n][r] + bias, 0.f);
    }
  }
}

// ---------------------------------------------------------------------------
extern "C" void kernel_launch(void* const* d_in, const int* in_sizes, int n_in,
                              void* d_out, int out_size, void* d_ws, size_t ws_size,
                              hipStream_t stream) {
  const float* x   = (const float*)d_in[0];
  const int*   ei  = (const int*)d_in[1];   // [2, E]: row0 = src(j), row1 = dst(i)
  const float* ea  = (const float*)d_in[2];
  const float* Wm1 = (const float*)d_in[3];  const float* bm1 = (const float*)d_in[4];
  const float* Wf1 = (const float*)d_in[5];  const float* bf1 = (const float*)d_in[6];
  const float* Wm2 = (const float*)d_in[7];  const float* bm2 = (const float*)d_in[8];
  const float* Wf2 = (const float*)d_in[9];  const float* bf2 = (const float*)d_in[10];
  const float* Wm3 = (const float*)d_in[11]; const float* bm3 = (const float*)d_in[12];
  const float* Wf3 = (const float*)d_in[13]; const float* bf3 = (const float*)d_in[14];
  const float* Wl  = (const float*)d_in[15]; const float* bl  = (const float*)d_in[16];
  (void)n_in; (void)out_size; (void)ws_size;

  const int Din = 64;
  const int N   = in_sizes[0] / Din;   // 20000
  const int E   = in_sizes[1] / 2;     // 320000

  char* ws = (char*)d_ws; size_t off = 0;
  auto walloc = [&](size_t bytes) -> void* {
    off = (off + 255) & ~(size_t)255;
    void* p = ws + off; off += bytes; return p;
  };
  float* sumb = (float*)walloc((size_t)N * ZDIM * 4);
  float* cnt  = (float*)walloc((size_t)N * 4);
  float* x1   = (float*)walloc((size_t)N * ZDIM * 4);
  float* x2   = (float*)walloc((size_t)N * ZDIM * 4);
  float* x3   = (float*)walloc((size_t)N * ZDIM * 4);
  unsigned short* fWm1 = (unsigned short*)walloc(192 * ZDIM * 2);
  unsigned short* fWf1 = (unsigned short*)walloc(192 * ZDIM * 2);
  unsigned short* fWm2 = (unsigned short*)walloc(384 * ZDIM * 2);
  unsigned short* fWf2 = (unsigned short*)walloc(256 * ZDIM * 2);
  unsigned short* fWm3 = (unsigned short*)walloc(384 * ZDIM * 2);
  unsigned short* fWf3 = (unsigned short*)walloc(256 * ZDIM * 2);
  unsigned short* fWl  = (unsigned short*)walloc(384 * ZDIM * 2);

  auto cdiv = [](int a, int b) { return (a + b - 1) / b; };
  const int B = 256, WPB = 8;  // 8 wave32 per block

  // Dynamic-LDS caps (non-stream host calls; deterministic, capture-safe).
  (void)hipFuncSetAttribute(reinterpret_cast<const void*>(&edge_msg_kernel<64>),
                            hipFuncAttributeMaxDynamicSharedMemorySize, 192 * 256);
  (void)hipFuncSetAttribute(reinterpret_cast<const void*>(&edge_msg_kernel<128>),
                            hipFuncAttributeMaxDynamicSharedMemorySize, 384 * 256);
  (void)hipFuncSetAttribute(reinterpret_cast<const void*>(&node_mlp_kernel<64>),
                            hipFuncAttributeMaxDynamicSharedMemorySize, 192 * 256);
  (void)hipFuncSetAttribute(reinterpret_cast<const void*>(&node_mlp_kernel<128>),
                            hipFuncAttributeMaxDynamicSharedMemorySize, 256 * 256);
  (void)hipFuncSetAttribute(reinterpret_cast<const void*>(&fusion_kernel),
                            hipFuncAttributeMaxDynamicSharedMemorySize, 384 * 256);

  // Swizzle weights into WMMA-B bf16 fragments (tiny; L2-resident thereafter).
  fragize_kernel<<<cdiv(192 * ZDIM, B), B, 0, stream>>>(Wm1, fWm1, 192);
  fragize_kernel<<<cdiv(192 * ZDIM, B), B, 0, stream>>>(Wf1, fWf1, 192);
  fragize_kernel<<<cdiv(384 * ZDIM, B), B, 0, stream>>>(Wm2, fWm2, 384);
  fragize_kernel<<<cdiv(256 * ZDIM, B), B, 0, stream>>>(Wf2, fWf2, 256);
  fragize_kernel<<<cdiv(384 * ZDIM, B), B, 0, stream>>>(Wm3, fWm3, 384);
  fragize_kernel<<<cdiv(256 * ZDIM, B), B, 0, stream>>>(Wf3, fWf3, 256);
  fragize_kernel<<<cdiv(384 * ZDIM, B), B, 0, stream>>>(Wl,  fWl,  384);

  // In-degree counts (shared by all three layers).
  zero_kernel<<<cdiv(N, B), B, 0, stream>>>(cnt, N);
  count_kernel<<<cdiv(E, B), B, 0, stream>>>(ei + E, cnt, E);

  const int eGrid = cdiv(cdiv(E, 16), WPB);
  const int nGrid = cdiv(cdiv(N, 16), WPB);

  // Layer 1 (D_IN = 64)
  zero_kernel<<<cdiv(N * ZDIM, B), B, 0, stream>>>(sumb, N * ZDIM);
  edge_msg_kernel<64><<<eGrid, B, 192 * 256, stream>>>(x, ei, ei + E, ea, fWm1, Wm1, bm1, sumb, E);
  node_mlp_kernel<64><<<nGrid, B, 192 * 256, stream>>>(x, sumb, cnt, fWf1, bf1, x1, N);

  // Layer 2 (Z = 128)
  zero_kernel<<<cdiv(N * ZDIM, B), B, 0, stream>>>(sumb, N * ZDIM);
  edge_msg_kernel<128><<<eGrid, B, 384 * 256, stream>>>(x1, ei, ei + E, ea, fWm2, Wm2, bm2, sumb, E);
  node_mlp_kernel<128><<<nGrid, B, 256 * 256, stream>>>(x1, sumb, cnt, fWf2, bf2, x2, N);

  // Layer 3
  zero_kernel<<<cdiv(N * ZDIM, B), B, 0, stream>>>(sumb, N * ZDIM);
  edge_msg_kernel<128><<<eGrid, B, 384 * 256, stream>>>(x2, ei, ei + E, ea, fWm3, Wm3, bm3, sumb, E);
  node_mlp_kernel<128><<<nGrid, B, 256 * 256, stream>>>(x2, sumb, cnt, fWf3, bf3, x3, N);

  // Fusion MLP
  fusion_kernel<<<nGrid, B, 384 * 256, stream>>>(x1, x2, x3, fWl, bl, (float*)d_out, N);
}
